// BiLSTM_68195490726439
// MI455X (gfx1250) — matrix operand order repaired and checked by reading, not compiled, not run on hardware
//
#include <hip/hip_runtime.h>
#include <hip/hip_bf16.h>
#include <math.h>

// ---------------------------------------------------------------------------
// BiLSTM (4 layers, bidirectional) + FC(20->2) + softmax for MI455X (gfx1250)
//
// Latency-bound recurrence (4 x 2048 dependent steps). Per layer: 32 work-
// groups (16 batch chunks x 2 dirs). Weights live in VGPRs (ISA B layout,
// preloaded once), h in double-buffered LDS (bf16, A layout friendly),
// c in VGPRs. Per step: prefetched global_load_b128 x-tiles, clause'd
// ds_load_b128 h-tiles, v_wmma_f32_16x16x32_bf16 chains (4 gate accums),
// branch-free v_tanh_f32 gate combine, and a split LDS-only barrier
// (s_wait_dscnt + s_barrier_signal/wait) so global stores never stall the
// recurrence.
// ---------------------------------------------------------------------------

typedef __attribute__((ext_vector_type(16))) __bf16 v16bf;
typedef __attribute__((ext_vector_type(8)))  __bf16 v8bf;
typedef __attribute__((ext_vector_type(8)))  float  v8f;

union V16 { v16bf v; v8bf h[2]; __bf16 e[16]; };
union V8  { v8f   v; float  e[8]; };

// hardware tanh on gfx1250; branch-free exp2/rcp fallback otherwise
__device__ __forceinline__ float fast_tanh(float x) {
#if __has_builtin(__builtin_amdgcn_tanhf)
  return __builtin_amdgcn_tanhf(x);
#else
  float e = __builtin_amdgcn_exp2f(x * 2.885390081777927f);  // exp(2x)
  return 1.0f - 2.0f * __builtin_amdgcn_rcpf(1.0f + e);
#endif
}
__device__ __forceinline__ float fast_sig(float x) {
  return 0.5f * fast_tanh(0.5f * x) + 0.5f;
}

// workgroup barrier that only drains LDS (DScnt), not global stores
__device__ __forceinline__ void wg_barrier_lds() {
  asm volatile("s_wait_dscnt 0x0\n\t"
               "s_barrier_signal -1\n\t"
               "s_barrier_wait -1" ::: "memory");
}

// f32 [N][3] -> bf16 [N][32] zero-padded
__global__ void cvt_pad_kernel(const float* __restrict__ in,
                               __bf16* __restrict__ out, int nrows) {
  int i = blockIdx.x * blockDim.x + threadIdx.x;
  if (i >= nrows * 32) return;
  int row = i >> 5, k = i & 31;
  out[i] = (k < 3) ? (__bf16)in[row * 3 + k] : (__bf16)0.0f;
}

// One bidirectional LSTM layer (one template instantiation per layer).
//  xin : [T][B][IPAD]  bf16, zero-padded input activations
//  hout: [T][B][2*HPAD] bf16, zero-padded output (fwd half | bwd half)
//  Input padding map: padded col k -> actual i = (k/SEGP)*SEGA + k%SEGP,
//  valid iff (k%SEGP) < SEGA.  (IPAD = nseg*SEGP, IACT = nseg*SEGA)
template <int H, int HPAD, int IACT, int IPAD, int SEGP, int SEGA>
__global__ void __launch_bounds__(2 * HPAD)
lstm_layer_tmpl(const __bf16* __restrict__ xin, __bf16* __restrict__ hout,
                const float* __restrict__ Wih0, const float* __restrict__ Whh0,
                const float* __restrict__ bih0, const float* __restrict__ bhh0,
                const float* __restrict__ Wih1, const float* __restrict__ Whh1,
                const float* __restrict__ bih1, const float* __restrict__ bhh1,
                int T, int B) {
  constexpr int HK  = (HPAD < 32) ? 32 : HPAD;  // K extent of recurrent GEMM
  constexpr int NXC = IPAD / 32;                // input K chunks
  constexpr int NHC = HK / 32;                  // hidden K chunks

  __shared__ __align__(16) __bf16 s_h[2][16 * HK];  // double-buffered h state

  const int dir   = blockIdx.y;
  const int brow0 = blockIdx.x * 16;
  const float* Wih = dir ? Wih1 : Wih0;
  const float* Whh = dir ? Whh1 : Whh0;
  const float* bih = dir ? bih1 : bih0;
  const float* bhh = dir ? bhh1 : bhh0;

  const int tid    = threadIdx.x;
  const int wave   = tid >> 5;
  const int lane   = tid & 31;
  const int mlo    = lane & 15;    // A row / B col / C col within a tile
  const int hiHalf = lane >> 4;    // lane half (ISA WMMA layouts)
  const int j      = 16 * wave + mlo;  // this wave's j column

  // zero both h buffers (covers HPAD..HK padding for the H=10 layer too)
  for (int i = tid; i < 2 * 16 * HK; i += blockDim.x)
    ((__bf16*)s_h)[i] = (__bf16)0.0f;

  // per-gate bias for this lane's column (0 on padded columns)
  float bias[4];
#pragma unroll
  for (int g = 0; g < 4; ++g)
    bias[g] = (j < H) ? (bih[g * H + j] + bhh[g * H + j]) : 0.0f;

  // ---- preload weights into registers, ISA B-matrix layout ----------------
  // B element e of chunk c: K = 32c + 8*hiHalf + (e<8 ? e : e+8), N-col = this lane's j.
  V16 Bih[NXC][4];
#pragma unroll
  for (int c = 0; c < NXC; ++c)
#pragma unroll
    for (int g = 0; g < 4; ++g)
#pragma unroll
      for (int e = 0; e < 16; ++e) {
        int k   = 32 * c + 8 * hiHalf + (e < 8 ? e : e + 8);  // padded input index
        int seg = k / SEGP;
        int ks  = k - seg * SEGP;
        bool ok = (j < H) && (ks < SEGA);
        Bih[c][g].e[e] =
            ok ? (__bf16)Wih[(g * H + j) * IACT + seg * SEGA + ks]
               : (__bf16)0.0f;
      }
  V16 Bhh[NHC][4];
#pragma unroll
  for (int c = 0; c < NHC; ++c)
#pragma unroll
    for (int g = 0; g < 4; ++g)
#pragma unroll
      for (int e = 0; e < 16; ++e) {
        int k   = 32 * c + 8 * hiHalf + (e < 8 ? e : e + 8);
        bool ok = (j < H) && (k < H);
        Bhh[c][g].e[e] =
            ok ? (__bf16)Whh[(g * H + j) * H + k] : (__bf16)0.0f;
      }

  float c0[8];
#pragma unroll
  for (int r = 0; r < 8; ++r) c0[r] = 0.0f;

  // incremental pointers (no per-step 64-bit address rebuilds)
  const ptrdiff_t xstep = (dir ? -1 : 1) * (ptrdiff_t)B * IPAD;
  const ptrdiff_t ostep = (dir ? -1 : 1) * (ptrdiff_t)B * (2 * HPAD);
  const int t0 = dir ? (T - 1) : 0;
  const __bf16* xptr =
      xin + ((size_t)t0 * B + brow0 + mlo) * IPAD + 8 * hiHalf;
  __bf16* optr =
      hout + ((size_t)t0 * B + brow0) * (2 * HPAD) + (size_t)dir * HPAD + j;

  // prologue: load the first x tile
  V16 Axc[NXC];
#pragma unroll
  for (int c = 0; c < NXC; ++c) {
    const v8bf* p = (const v8bf*)(xptr + 32 * c);
    Axc[c].h[0] = p[0];  // K = 32c + 8*hiHalf + 0..7
    Axc[c].h[1] = p[2];  // K = 32c + 8*hiHalf + 16..23
  }

  wg_barrier_lds();

  for (int t = 0; t < T; ++t) {
    const int rb = t & 1;   // read h buffer
    const int wb = rb ^ 1;  // write h buffer

    // prefetch next step's x tile (clamped on the last iteration)
    const __bf16* xnext = xptr + ((t + 1 < T) ? xstep : (ptrdiff_t)0);
    V16 Axn[NXC];
#pragma unroll
    for (int c = 0; c < NXC; ++c) {
      const v8bf* p = (const v8bf*)(xnext + 32 * c);
      Axn[c].h[0] = p[0];
      Axn[c].h[1] = p[2];
    }

    V8 acc[4];
#pragma unroll
    for (int g = 0; g < 4; ++g)
#pragma unroll
      for (int r = 0; r < 8; ++r) acc[g].e[r] = bias[g];

    // A_h: clause of aligned ds_load_b128, single dscnt wait
    V16 Ah[NHC];
#pragma unroll
    for (int c = 0; c < NHC; ++c) {
      const v8bf* q = (const v8bf*)&s_h[rb][mlo * HK + 32 * c + 8 * hiHalf];
      Ah[c].h[0] = q[0];
      Ah[c].h[1] = q[2];
    }

    // recurrent chain
#pragma unroll
    for (int c = 0; c < NHC; ++c)
#pragma unroll
      for (int g = 0; g < 4; ++g)
        acc[g].v = __builtin_amdgcn_wmma_f32_16x16x32_bf16(
            false, Ah[c].v, false, Bhh[c][g].v, (short)0, acc[g].v, false,
            false);

    // input chain (current step's prefetched tile)
#pragma unroll
    for (int c = 0; c < NXC; ++c)
#pragma unroll
      for (int g = 0; g < 4; ++g)
        acc[g].v = __builtin_amdgcn_wmma_f32_16x16x32_bf16(
            false, Axc[c].v, false, Bih[c][g].v, (short)0, acc[g].v, false,
            false);

    // ---- gate combine: branch-free v_tanh_f32, all in registers ---------
    __bf16* hwr = &s_h[wb][0];
#pragma unroll
    for (int r = 0; r < 8; ++r) {
      int   m  = r + 8 * hiHalf;
      float ci = fast_sig(acc[0].e[r]);
      float cf = fast_sig(acc[1].e[r]);
      float cg = fast_tanh(acc[2].e[r]);
      float co = fast_sig(acc[3].e[r]);
      float c  = cf * c0[r] + ci * cg;
      float h  = co * fast_tanh(c);
      c0[r]    = c;
      __bf16 hb = (__bf16)h;
      hwr[m * HK + j]              = hb;  // next step's A operand
      optr[(size_t)m * (2 * HPAD)] = hb;  // layer output (padded cols are 0)
    }

    // LDS-only barrier: drains DScnt, NOT the global output stores
    wg_barrier_lds();

    xptr = xnext;
    optr += ostep;
#pragma unroll
    for (int c = 0; c < NXC; ++c) Axc[c] = Axn[c];
  }
}

// FC(20->2) + 2-way softmax over padded final activations [N][32]
// (fwd h at cols 0..9, bwd h at cols 16..25)
__global__ void fc_softmax_kernel(const __bf16* __restrict__ hin,
                                  const float* __restrict__ w,   // [2][20]
                                  const float* __restrict__ b,   // [2]
                                  float* __restrict__ out, int N) {
  int i = blockIdx.x * blockDim.x + threadIdx.x;
  if (i >= N) return;
  const __bf16* hp = hin + (size_t)i * 32;
  float l0 = b[0], l1 = b[1];
#pragma unroll
  for (int k = 0; k < 10; ++k) {
    float hv = (float)hp[k];
    l0 += hv * w[k];
    l1 += hv * w[20 + k];
  }
#pragma unroll
  for (int k = 0; k < 10; ++k) {
    float hv = (float)hp[16 + k];
    l0 += hv * w[10 + k];
    l1 += hv * w[30 + k];
  }
  float m  = fmaxf(l0, l1);
  float e0 = __expf(l0 - m), e1 = __expf(l1 - m);
  float s  = e0 + e1;
  out[2 * i + 0] = e0 / s;
  out[2 * i + 1] = e1 / s;
}

// ---------------------------------------------------------------------------

extern "C" void kernel_launch(void* const* d_in, const int* in_sizes, int n_in,
                              void* d_out, int out_size, void* d_ws, size_t ws_size,
                              hipStream_t stream) {
  (void)in_sizes; (void)out_size; (void)ws_size;
  const int T = 2048, B = 256;

  const float* x = (const float*)d_in[0];
  const float* P[4][2][4];
  for (int li = 0; li < 4; ++li)
    for (int d = 0; d < 2; ++d)
      for (int k = 0; k < 4; ++k)
        P[li][d][k] = (const float*)d_in[1 + (li * 2 + d) * 4 + k];
  const float* fcw = (const float*)d_in[n_in - 2];
  const float* fcb = (const float*)d_in[n_in - 1];

  // workspace carve: zero-padded bf16 activation buffers
  char*  ws  = (char*)d_ws;
  size_t off = 0;
  auto carve = [&](size_t bytes) -> void* {
    void* p = ws + off;
    off += (bytes + 255) & ~(size_t)255;
    return p;
  };
  const size_t TB = (size_t)T * B;
  __bf16* xpad = (__bf16*)carve(TB * 32 * sizeof(__bf16));   // [TB][32]
  __bf16* act0 = (__bf16*)carve(TB * 128 * sizeof(__bf16));  // [TB][2*64]
  __bf16* act1 = (__bf16*)carve(TB * 64 * sizeof(__bf16));   // [TB][2*32]
  __bf16* act2 = (__bf16*)carve(TB * 64 * sizeof(__bf16));   // [TB][2*32]
  __bf16* act3 = (__bf16*)carve(TB * 32 * sizeof(__bf16));   // [TB][2*16]

  // 1) input f32 [TB][3] -> bf16 [TB][32] zero-padded
  {
    int n = (int)(TB * 32);
    cvt_pad_kernel<<<(n + 255) / 256, 256, 0, stream>>>(x, xpad, (int)TB);
  }

  dim3 grid(B / 16, 2);

  // 2) 4 BiLSTM layers (template per layer: H, HPAD, IACT, IPAD, SEGP, SEGA)
  lstm_layer_tmpl<50, 64, 3, 32, 32, 3><<<grid, 128, 0, stream>>>(
      xpad, act0, P[0][0][0], P[0][0][1], P[0][0][2], P[0][0][3],
      P[0][1][0], P[0][1][1], P[0][1][2], P[0][1][3], T, B);

  lstm_layer_tmpl<25, 32, 100, 128, 64, 50><<<grid, 64, 0, stream>>>(
      act0, act1, P[1][0][0], P[1][0][1], P[1][0][2], P[1][0][3],
      P[1][1][0], P[1][1][1], P[1][1][2], P[1][1][3], T, B);

  lstm_layer_tmpl<25, 32, 50, 64, 32, 25><<<grid, 64, 0, stream>>>(
      act1, act2, P[2][0][0], P[2][0][1], P[2][0][2], P[2][0][3],
      P[2][1][0], P[2][1][1], P[2][1][2], P[2][1][3], T, B);

  lstm_layer_tmpl<10, 16, 50, 64, 32, 25><<<grid, 32, 0, stream>>>(
      act2, act3, P[3][0][0], P[3][0][1], P[3][0][2], P[3][0][3],
      P[3][1][0], P[3][1][1], P[3][1][2], P[3][1][3], T, B);

  // 3) FC + softmax
  {
    int n = (int)TB;
    fc_softmax_kernel<<<(n + 255) / 256, 256, 0, stream>>>(
        act3, fcw, fcb, (float*)d_out, n);
  }
}
